// TopKSparseAutoencoder_51917564674099
// MI455X (gfx1250) — matrix operand order repaired and checked by reading, not compiled, not run on hardware
//
#include <hip/hip_runtime.h>
#include <hip/hip_bf16.h>

#define BATCH   16384
#define IN_DIM  768
#define LATENT  12288
#define TOPK    32

typedef __attribute__((ext_vector_type(16))) __bf16 v16bf;
typedef __attribute__((ext_vector_type(8)))  __bf16 bf8;
typedef __attribute__((ext_vector_type(8)))  float  v8f;

// ---------------------------------------------------------------------------
// Kernel 1: encoder GEMM  z[b,l] = sum_i x[b,i]*enc_w[l,i] + enc_b[l]
// bf16 WMMA (16x16x32), f32 accumulate. WG = 256 thr = 8 waves -> 128x128 tile.
// Wave grid 2(M) x 4(N); each wave: 4x2 wmma tiles = 64x32 outputs.
// ---------------------------------------------------------------------------

#define KTILE   32
#define NSTAGE  (IN_DIM / KTILE)      // 24
#define LDS_ROW 40                    // 32 + 8 pad bf16 elems -> 80B row stride

__device__ __forceinline__ void stage_regs(__bf16* dst,
                                           float4 f0, float4 f1,
                                           float4 f2, float4 f3)
{
    bf8 lo, hi;
    lo[0]=(__bf16)f0.x; lo[1]=(__bf16)f0.y; lo[2]=(__bf16)f0.z; lo[3]=(__bf16)f0.w;
    lo[4]=(__bf16)f1.x; lo[5]=(__bf16)f1.y; lo[6]=(__bf16)f1.z; lo[7]=(__bf16)f1.w;
    hi[0]=(__bf16)f2.x; hi[1]=(__bf16)f2.y; hi[2]=(__bf16)f2.z; hi[3]=(__bf16)f2.w;
    hi[4]=(__bf16)f3.x; hi[5]=(__bf16)f3.y; hi[6]=(__bf16)f3.z; hi[7]=(__bf16)f3.w;
    *(bf8*)dst       = lo;   // ds_store_b128
    *(bf8*)(dst + 8) = hi;   // ds_store_b128
}

// A fragment (16x32 bf16): lane holds row m=l15; K chunks [koff..koff+7] and
// [16+koff..23+koff], koff = (lane>=16 ? 8 : 0).
__device__ __forceinline__ v16bf load_frag_a(const __bf16* s, int row, int koff)
{
    const __bf16* p = s + row * LDS_ROW;
    bf8 lo = *(const bf8*)(p + koff);
    bf8 hi = *(const bf8*)(p + 16 + koff);
    return __builtin_shufflevector(lo, hi, 0,1,2,3,4,5,6,7,8,9,10,11,12,13,14,15);
}

// B fragment (32x16 bf16): lane holds col n=l15; K chunks [koff..koff+7] and
// [16+koff..23+koff], koff = (lane>=16 ? 8 : 0). B[k,n] = enc_w[n,k] -> row-contiguous in enc_w.
__device__ __forceinline__ v16bf load_frag_b(const __bf16* s, int row, int koff)
{
    const __bf16* p = s + row * LDS_ROW + koff;
    bf8 lo = *(const bf8*)(p);
    bf8 hi = *(const bf8*)(p + 16);
    return __builtin_shufflevector(lo, hi, 0,1,2,3,4,5,6,7,8,9,10,11,12,13,14,15);
}

__launch_bounds__(256)
__global__ void enc_gemm_kernel(const float* __restrict__ x,
                                const float* __restrict__ w,
                                const float* __restrict__ bias,
                                float* __restrict__ z)
{
    __shared__ alignas(16) __bf16 sA[2][128][LDS_ROW];   // 20 KB
    __shared__ alignas(16) __bf16 sB[2][128][LDS_ROW];   // 20 KB

    const int t    = threadIdx.x;
    const int lane = t & 31;
    const int wv   = t >> 5;        // 0..7
    const int wm   = wv >> 2;       // 0..1
    const int wn   = wv & 3;        // 0..3
    const int l15  = lane & 15;
    const int lh   = lane >> 4;     // 0/1

    const int Mbase = blockIdx.y * 128;
    const int Nbase = blockIdx.x * 128;

    // global->LDS staging assignment: thread t loads 16 f32 of one row
    const int lr = t >> 1;          // 0..127
    const int lc = (t & 1) << 4;    // 0 / 16

    const float* gA = x + (size_t)(Mbase + lr) * IN_DIM + lc;
    const float* gB = w + (size_t)(Nbase + lr) * IN_DIM + lc;

    {   // stage 0 (direct indexed stores -> ds_store_b128, addrspace(3) kept)
        const float4* pA = (const float4*)gA;
        const float4* pB = (const float4*)gB;
        stage_regs(&sA[0][lr][lc], pA[0], pA[1], pA[2], pA[3]);
        stage_regs(&sB[0][lr][lc], pB[0], pB[1], pB[2], pB[3]);
    }
    __syncthreads();

    v8f acc[4][2] = {};

    const int koffA = lh << 3;      // 0 / 8
    const int koffB = lh << 3;      // 0 / 8

    for (int kt = 0; kt < NSTAGE; ++kt) {
        const int cur = kt & 1;
        const int nxt = cur ^ 1;
        const __bf16* cA = &sA[cur][0][0];
        const __bf16* cB = &sB[cur][0][0];

        // prefetch next K-slab into registers (overlaps with wmma below)
        float4 a0, a1, a2, a3, b0, b1, b2, b3;
        const bool more = (kt + 1) < NSTAGE;
        if (more) {
            const float4* pA = (const float4*)(gA + (kt + 1) * KTILE);
            const float4* pB = (const float4*)(gB + (kt + 1) * KTILE);
            a0 = pA[0]; a1 = pA[1]; a2 = pA[2]; a3 = pA[3];
            b0 = pB[0]; b1 = pB[1]; b2 = pB[2]; b3 = pB[3];
        }

        v16bf af[4];
        #pragma unroll
        for (int mt = 0; mt < 4; ++mt)
            af[mt] = load_frag_a(cA, wm * 64 + mt * 16 + l15, koffA);
        v16bf bfr[2];
        #pragma unroll
        for (int nt = 0; nt < 2; ++nt)
            bfr[nt] = load_frag_b(cB, wn * 32 + nt * 16 + l15, koffB);

        #pragma unroll
        for (int mt = 0; mt < 4; ++mt)
            #pragma unroll
            for (int nt = 0; nt < 2; ++nt)
                acc[mt][nt] = __builtin_amdgcn_wmma_f32_16x16x32_bf16(
                    false, af[mt], false, bfr[nt],
                    (short)0, acc[mt][nt], false, false);

        if (more) {
            // direct indexed -> stays addrspace(3) -> ds_store_b128 (no flat)
            stage_regs(&sA[nxt][lr][lc], a0, a1, a2, a3);
            stage_regs(&sB[nxt][lr][lc], b0, b1, b2, b3);
        }
        __syncthreads();
    }

    // Epilogue: C/D layout -> VGPR r: lanes 0-15 M=r, lanes 16-31 M=8+r, N=l15
    #pragma unroll
    for (int nt = 0; nt < 2; ++nt) {
        const int   n  = Nbase + wn * 32 + nt * 16 + l15;
        const float bv = bias[n];
        #pragma unroll
        for (int mt = 0; mt < 4; ++mt) {
            const int m0 = Mbase + wm * 64 + mt * 16 + (lh << 3);
            #pragma unroll
            for (int r = 0; r < 8; ++r)
                z[(size_t)(m0 + r) * LATENT + n] = acc[mt][nt][r] + bv;
        }
    }
}

// ---------------------------------------------------------------------------
// Kernel 2: per-row kth-largest via radix bisection on order-preserving keys,
// in-place relu-threshold of z, compaction, sparse decode from L2-resident
// dec_w (37.7 MB << 192 MB L2).
// ---------------------------------------------------------------------------

#define PER (LATENT / 256)   // 48 elements per thread, held in registers

__device__ __forceinline__ unsigned f2k(float f)
{
    unsigned u = __float_as_uint(f);
    return (u & 0x80000000u) ? ~u : (u | 0x80000000u);
}
__device__ __forceinline__ float k2f(unsigned k)
{
    unsigned u = (k & 0x80000000u) ? (k & 0x7fffffffu) : ~k;
    return __uint_as_float(u);
}

__launch_bounds__(256)
__global__ void topk_decode_kernel(const float* __restrict__ dec_w,
                                   const float* __restrict__ dec_b,
                                   float* __restrict__ z,
                                   float* __restrict__ xr)
{
    __shared__ int   s_cnt[32];     // one counter per radix bit -> 1 barrier/iter
    __shared__ int   s_nnz;
    __shared__ int   s_idx[TOPK];
    __shared__ float s_val[TOPK];

    const int t = threadIdx.x;
    const int b = blockIdx.x;
    float* zrow = z + (size_t)b * LATENT;

    if (t < 32) s_cnt[t] = 0;
    if (t == 0) s_nnz = 0;

    // load row, transform to monotone uint keys, keep in registers
    unsigned kreg[PER];
    #pragma unroll
    for (int i = 0; i < PER; ++i)
        kreg[i] = f2k(zrow[t + (i << 8)]);
    __syncthreads();

    // greedy MSB-first search: largest key T with count(key >= T) >= TOPK
    unsigned prefix = 0u;
    for (int bit = 31; bit >= 0; --bit) {
        const unsigned cand = prefix | (1u << bit);
        int c = 0;
        #pragma unroll
        for (int i = 0; i < PER; ++i)
            c += (kreg[i] >= cand) ? 1 : 0;
        #pragma unroll
        for (int off = 16; off > 0; off >>= 1)
            c += __shfl_xor(c, off, 32);
        if ((t & 31) == 0) atomicAdd(&s_cnt[bit], c);
        __syncthreads();
        if (s_cnt[bit] >= TOPK) prefix = cand;
    }

    const float tf = k2f(prefix);   // kth largest value (= reference thresh)

    // z_sparse = relu(z - thresh), written in place; compact the (<=31)
    // strictly-positive survivors (ties at thresh relu to exactly 0).
    #pragma unroll
    for (int i = 0; i < PER; ++i) {
        const int   j = t + (i << 8);
        const float v = k2f(kreg[i]);
        const float o = fmaxf(v - tf, 0.0f);
        zrow[j] = o;
        if (o > 0.0f) {
            const int p = atomicAdd(&s_nnz, 1);
            if (p < TOPK) { s_idx[p] = j; s_val[p] = o; }
        }
    }
    __syncthreads();

    int cnt = s_nnz;
    if (cnt > TOPK) cnt = TOPK;

    // sparse decode: x_recon[b,i] = dec_b[i] + sum_c val_c * dec_w[i, idx_c]
    for (int i = t; i < IN_DIM; i += 256) {
        float acc = dec_b[i];
        const float* wrow = dec_w + (size_t)i * LATENT;
        for (int c = 0; c < cnt; ++c)
            acc += s_val[c] * wrow[s_idx[c]];
        xr[(size_t)b * IN_DIM + i] = acc;
    }
}

// ---------------------------------------------------------------------------

extern "C" void kernel_launch(void* const* d_in, const int* in_sizes, int n_in,
                              void* d_out, int out_size, void* d_ws, size_t ws_size,
                              hipStream_t stream)
{
    const float* x     = (const float*)d_in[0];
    const float* enc_w = (const float*)d_in[1];
    const float* enc_b = (const float*)d_in[2];
    const float* dec_w = (const float*)d_in[3];
    const float* dec_b = (const float*)d_in[4];
    (void)in_sizes; (void)n_in; (void)out_size; (void)d_ws; (void)ws_size;

    float* xr = (float*)d_out;                               // [BATCH, IN_DIM]
    float* z  = (float*)d_out + (size_t)BATCH * IN_DIM;      // [BATCH, LATENT]

    dim3 g1(LATENT / 128, BATCH / 128);                      // 96 x 128 tiles
    enc_gemm_kernel<<<g1, 256, 0, stream>>>(x, enc_w, enc_b, z);
    topk_decode_kernel<<<BATCH, 256, 0, stream>>>(dec_w, dec_b, z, xr);
}
